// testModel_59047210385967
// MI455X (gfx1250) — matrix operand order repaired and checked by reading, not compiled, not run on hardware
//
#include <hip/hip_runtime.h>
#include <hip/hip_bf16.h>

typedef __attribute__((ext_vector_type(16))) _Float16 v16h;
typedef __attribute__((ext_vector_type(8)))  float    v8f;

#define Bsz 16
#define Lsz 128
#define Dsz 128
#define OUTsz 10

// ---------------------------------------------------------------------------
// K1: fold (D,D,D).sum(axis=0): eff[r*D+c] = sum_t W[(t*D+r)*D + c]
// 16384 outputs, one thread each; consecutive threads -> coalesced reads.
// ---------------------------------------------------------------------------
__global__ void fold_w1_kernel(const float* __restrict__ W, float* __restrict__ eff) {
    int idx = blockIdx.x * blockDim.x + threadIdx.x;   // r*D + c, 0..16383
    if (idx >= Dsz * Dsz) return;
    float acc = 0.0f;
    #pragma unroll 4
    for (int t = 0; t < Dsz; ++t)
        acc += W[t * (Dsz * Dsz) + idx];
    eff[idx] = acc;
}

// ---------------------------------------------------------------------------
// K2: embedding gather: e[(b*L+l)*D+d] = word_emb[word*D+d] + pos_emb[pos*D+d]
// ---------------------------------------------------------------------------
__global__ void embed_kernel(const int* __restrict__ word_data,
                             const int* __restrict__ position_data,
                             const float* __restrict__ word_emb,
                             const float* __restrict__ pos_emb,
                             float* __restrict__ e) {
    int idx = blockIdx.x * blockDim.x + threadIdx.x;   // over B*L*D
    int d   = idx & (Dsz - 1);
    int bl  = idx >> 7;                                 // b*L + l
    int w   = word_data[bl];
    int pp  = position_data[bl];
    e[idx] = word_emb[w * Dsz + d] + pos_emb[pp * Dsz + d];
}

// ---------------------------------------------------------------------------
// K3: generic GEMM  C(M,128) = act( A(M,128) @ W(128,128) + bias + add )
// One 16x16 C tile per wave via v_wmma_f32_16x16x32_f16 (f32 accumulate).
// Block = 256 threads = 8 waves covering the full N=128; grid = M/16.
// ---------------------------------------------------------------------------
__global__ void gemm128_wmma_kernel(const float* __restrict__ A,
                                    const float* __restrict__ W,
                                    const float* __restrict__ bias, int has_bias,
                                    const float* __restrict__ add,  int has_add,
                                    int do_relu,
                                    float* __restrict__ C) {
    const int lane = threadIdx.x & 31;
    const int wave = threadIdx.x >> 5;          // 0..7 -> N tile
    const int m0   = blockIdx.x * 16;
    const int n0   = wave * 16;

    v8f acc = {};
    // A-fragment K sub-block base per half-wave (ISA 16-bit A 16x32 layout)
    const int akb = (lane < 16) ? 0 : 8;
    const int bkb = (lane < 16) ? 0 : 8;
    const int am  = m0 + (lane & 15);
    const int bn  = (lane & 15);

    #pragma unroll
    for (int k0 = 0; k0 < Dsz; k0 += 32) {
        v16h af, bf;
        const float* arow = A + am * Dsz + k0;
        #pragma unroll
        for (int t = 0; t < 8; ++t) {
            af[t]     = (_Float16)arow[akb + t];        // K = k0 + akb + t
            af[8 + t] = (_Float16)arow[16 + akb + t];   // K = k0 + 16 + akb + t
        }
        const float* bcol = W + (k0 + bkb) * Dsz + bn;
        #pragma unroll
        for (int t = 0; t < 8; ++t) {
            bf[t]     = (_Float16)bcol[t * Dsz];        // B[k0+bkb+t][bn]
            bf[8 + t] = (_Float16)bcol[(16 + t) * Dsz]; // B[k0+16+bkb+t][bn]
        }
        acc = __builtin_amdgcn_wmma_f32_16x16x32_f16(
            /*neg_a=*/false, af, /*neg_b=*/false, bf,
            /*c_mod=*/(short)0, acc, /*reuse_a=*/false, /*reuse_b=*/false);
    }

    // C/D layout: VGPR v -> row (v or v+8), column = lane&15
    const int n = n0 + (lane & 15);
    const int mbase = m0 + ((lane < 16) ? 0 : 8);
    const float bv = has_bias ? bias[n] : 0.0f;
    #pragma unroll
    for (int v = 0; v < 8; ++v) {
        float x = acc[v] + bv;
        if (has_add) x += add[(mbase + v) * Dsz + n];
        if (do_relu) x = fmaxf(x, 0.0f);
        C[(mbase + v) * Dsz + n] = x;
    }
}

// ---------------------------------------------------------------------------
// K4: sent[b,i,d] = sum_j tanh(r[b,i,d]+s[b,j,d]) * (p[b,i,d]+q[b,j,d])
// grid = B * 4 (i-tiles of 32); block = 256 (d = tid&127, half = tid>>7).
// Each thread: 1 column d, 16 rows i in registers; s/q staged in 32-row
// LDS tiles, each LDS load amortized over 16 tanh-FMAs.
// ---------------------------------------------------------------------------
__device__ __forceinline__ float fast_tanh(float x) {
    // tanh(x) = 1 - 2/(exp(2x)+1); exp(2x) = exp2(2*log2(e)*x)
    float t = __builtin_amdgcn_exp2f(x * 2.8853900817779268f);
    return 1.0f - 2.0f * __builtin_amdgcn_rcpf(t + 1.0f);
}

#define IPT 16   // i-rows per thread
__global__ void sent_kernel(const float* __restrict__ p,
                            const float* __restrict__ q,
                            const float* __restrict__ r,
                            const float* __restrict__ s,
                            float* __restrict__ sent) {
    __shared__ float s_sh[32 * Dsz];
    __shared__ float q_sh[32 * Dsz];

    const int b  = blockIdx.x >> 2;
    const int it = blockIdx.x & 3;
    const int d  = threadIdx.x & (Dsz - 1);
    const int i0 = it * 32 + (threadIdx.x >> 7) * IPT;

    float rreg[IPT], preg[IPT], acc[IPT];
    #pragma unroll
    for (int ii = 0; ii < IPT; ++ii) {
        int row = (b * Lsz + i0 + ii) * Dsz + d;
        rreg[ii] = r[row];
        preg[ii] = p[row];
        acc[ii]  = 0.0f;
    }

    for (int jt = 0; jt < Lsz; jt += 32) {
        __syncthreads();
        #pragma unroll
        for (int u = 0; u < (32 * Dsz) / 256; ++u) {
            int idx = u * 256 + threadIdx.x;            // j*128 + d within tile
            int g   = (b * Lsz + jt + (idx >> 7)) * Dsz + (idx & (Dsz - 1));
            s_sh[idx] = s[g];
            q_sh[idx] = q[g];
        }
        __syncthreads();
        #pragma unroll 4
        for (int jj = 0; jj < 32; ++jj) {
            float sv = s_sh[jj * Dsz + d];
            float qv = q_sh[jj * Dsz + d];
            #pragma unroll
            for (int ii = 0; ii < IPT; ++ii)
                acc[ii] += fast_tanh(rreg[ii] + sv) * (preg[ii] + qv);
        }
    }

    #pragma unroll
    for (int ii = 0; ii < IPT; ++ii)
        sent[(b * Lsz + i0 + ii) * Dsz + d] = acc[ii];
}

// ---------------------------------------------------------------------------
// K5: out[b,o] = sum_t h[b, t] * W_d5[t, o];  one block per (b,o).
// ---------------------------------------------------------------------------
__global__ void final_gemv_kernel(const float* __restrict__ h,
                                  const float* __restrict__ W_d5,
                                  float* __restrict__ out) {
    __shared__ float red[256];
    const int b = blockIdx.x / OUTsz;
    const int o = blockIdx.x % OUTsz;
    const int T = Lsz * Dsz;                     // 16384
    float acc = 0.0f;
    for (int t = threadIdx.x; t < T; t += 256)
        acc += h[b * T + t] * W_d5[t * OUTsz + o];
    red[threadIdx.x] = acc;
    __syncthreads();
    #pragma unroll
    for (int st = 128; st > 0; st >>= 1) {
        if ((int)threadIdx.x < st) red[threadIdx.x] += red[threadIdx.x + st];
        __syncthreads();
    }
    if (threadIdx.x == 0) out[b * OUTsz + o] = red[0];
}

// ---------------------------------------------------------------------------
extern "C" void kernel_launch(void* const* d_in, const int* in_sizes, int n_in,
                              void* d_out, int out_size, void* d_ws, size_t ws_size,
                              hipStream_t stream) {
    (void)in_sizes; (void)n_in; (void)out_size; (void)ws_size;
    const int*   word_data     = (const int*)d_in[0];
    const int*   position_data = (const int*)d_in[1];
    // d_in[2] = sentence_length: unused by the reference math
    const float* word_emb = (const float*)d_in[3];
    const float* pos_emb  = (const float*)d_in[4];
    const float* W_dw1    = (const float*)d_in[5];
    const float* b_dw1    = (const float*)d_in[6];
    const float* W_dw2    = (const float*)d_in[7];
    const float* b_dw2    = (const float*)d_in[8];
    const float* W_du1    = (const float*)d_in[9];
    const float* b_du1    = (const float*)d_in[10];
    const float* W_du2    = (const float*)d_in[11];
    const float* b_du2    = (const float*)d_in[12];
    const float* W_d4     = (const float*)d_in[13];
    const float* b_d4     = (const float*)d_in[14];
    const float* W_d5     = (const float*)d_in[15];
    float* out = (float*)d_out;

    const int M  = Bsz * Lsz;          // 2048
    const int MD = M * Dsz;            // 262144 floats per activation
    float* ws     = (float*)d_ws;
    float* e      = ws;                // MD
    float* p      = ws + 1 * MD;
    float* q      = ws + 2 * MD;
    float* t      = ws + 3 * MD;
    float* r      = ws + 4 * MD;
    float* s      = ws + 5 * MD;
    float* sent   = ws + 6 * MD;
    float* h      = ws + 7 * MD;
    float* W1eff  = ws + 8 * MD;             // 16384
    float* Wu1eff = ws + 8 * MD + Dsz * Dsz; // 16384

    // Fold (D,D,D).sum(0) for W_dw1 and W_du1
    fold_w1_kernel<<<(Dsz * Dsz + 255) / 256, 256, 0, stream>>>(W_dw1, W1eff);
    fold_w1_kernel<<<(Dsz * Dsz + 255) / 256, 256, 0, stream>>>(W_du1, Wu1eff);

    // Embedding gather
    embed_kernel<<<MD / 256, 256, 0, stream>>>(word_data, position_data,
                                               word_emb, pos_emb, e);

    // p = e @ W1eff + b_dw1 ; q = e @ W_dw2 + b_dw2 ; t = e @ Wu1eff + b_du1
    gemm128_wmma_kernel<<<M / 16, 256, 0, stream>>>(e, W1eff,  b_dw1, 1, nullptr, 0, 0, p);
    gemm128_wmma_kernel<<<M / 16, 256, 0, stream>>>(e, W_dw2,  b_dw2, 1, nullptr, 0, 0, q);
    gemm128_wmma_kernel<<<M / 16, 256, 0, stream>>>(e, Wu1eff, b_du1, 1, nullptr, 0, 0, t);
    // r = t + p @ W_du2 ; s = q @ W_du2 + b_du2
    gemm128_wmma_kernel<<<M / 16, 256, 0, stream>>>(p, W_du2, nullptr, 0, t, 1, 0, r);
    gemm128_wmma_kernel<<<M / 16, 256, 0, stream>>>(q, W_du2, b_du2, 1, nullptr, 0, 0, s);

    // sent[b,i,d] = sum_j tanh(r_i + s_j) * (p_i + q_j)
    sent_kernel<<<Bsz * 4, 256, 0, stream>>>(p, q, r, s, sent);

    // h = relu(sent @ W_d4 + b_d4)
    gemm128_wmma_kernel<<<M / 16, 256, 0, stream>>>(sent, W_d4, b_d4, 1, nullptr, 0, 1, h);

    // out = h.reshape(B, L*D) @ W_d5
    final_gemv_kernel<<<Bsz * OUTsz, 256, 0, stream>>>(h, W_d5, out);
}
